// Net_15762529976716
// MI455X (gfx1250) — matrix operand order repaired and checked by reading, not compiled, not run on hardware
//
#include <hip/hip_runtime.h>
#include <hip/hip_bf16.h>

// ---------- WMMA types (CDNA5 / gfx1250, wave32) ----------
typedef __attribute__((ext_vector_type(16))) __bf16 v16bf;
typedef __attribute__((ext_vector_type(8)))  float  v8f;

#define TB 256

// ---------- helpers ----------
__device__ __forceinline__ void atomicMaxF(float* addr, float v) {
    // sign-split trick: monotone in int-space for >=0, reverse-monotone in uint-space for <0
    if (v >= 0.0f) atomicMax((int*)addr, __float_as_int(v));
    else           atomicMin((unsigned int*)addr, __float_as_uint(v));
}

__global__ void fill_f32(float* __restrict__ p, float v, long long n) {
    long long i = (long long)blockIdx.x * blockDim.x + threadIdx.x;
    if (i < n) p[i] = v;
}

// ---------- bf16 WMMA GEMM:  C[N,64] = f(A[N,KTOT]) @ W[KTOT,64] ----------
// MODE 0: f = identity (layer-1: A = x)
// MODE 1: f = elu(A + bias)  (layer-2: A = agg1, bias = b1) -- fused activation
//
// 8 waves/block; each wave computes 16 rows x 64 cols (4 WMMA accumulators
// sharing one A fragment) -> block covers 128 rows, x is read once from HBM.
// A fragment loaded as 4x global_load_b128 (ISA 16-bit A layout = two
// contiguous 8-float runs per lane). W staged to LDS as bf16 in
// fragment-ready layout (<=32 KB per stage), inner loop reads it with
// 2x ds_load_b128 per tile. Load rows clamped / store rows guarded for the
// ragged tail so EXEC is all-ones at every WMMA.
template<int KTOT, int MODE>
__global__ __launch_bounds__(TB) void gemm_wmma_n64(
        const float* __restrict__ A,
        const float* __restrict__ Wm,
        const float* __restrict__ bias,
        float* __restrict__ C, long long Nrows) {
    constexpr int NKT = KTOT / 32;              // k-tiles total
    constexpr int SKT = (NKT > 8) ? 8 : NKT;    // k-tiles staged per round (<=32 KB)
    __shared__ __bf16 Wl[SKT * 4 * 512];        // [ktLocal][ntile][lane][e]

    const int lane = threadIdx.x & 31;
    const int wave = threadIdx.x >> 5;
    const int half = lane >> 4;                 // 0: lanes 0-15, 1: lanes 16-31
    const int l16  = lane & 15;

    const long long m0 = (long long)blockIdx.x * 128 + (long long)wave * 16;
    long long row = m0 + l16;
    if (row >= Nrows) row = Nrows - 1;          // clamp: OOB outputs discarded
    const float* __restrict__ Arow = A + (size_t)row * KTOT;

    v8f acc[4] = {};

    for (int ks = 0; ks < NKT; ks += SKT) {
        __syncthreads();
        // ---- stage W -> LDS (bf16, fragment-ready layout) ----
        for (int idx = threadIdx.x; idx < SKT * 2048; idx += TB) {
            const int e  = idx & 15;
            const int ln = (idx >> 4) & 31;
            const int nt = (idx >> 9) & 3;
            const int kt = (idx >> 11) + ks;
            const int k  = kt * 32 + (e & 8) * 2 + (ln >> 4) * 8 + (e & 7);
            const int c  = nt * 16 + (ln & 15);
            Wl[idx] = (__bf16)Wm[(size_t)k * 64 + c];
        }
        __syncthreads();

        for (int kk = 0; kk < SKT; ++kk) {
            const int k0 = (ks + kk) * 32;
            // ---- A fragment: two contiguous 8-float runs per lane ----
            const float* ap = Arow + k0 + half * 8;
            const float4 f0 = ((const float4*)ap)[0];
            const float4 f1 = ((const float4*)ap)[1];
            const float4 f2 = ((const float4*)(ap + 16))[0];
            const float4 f3 = ((const float4*)(ap + 16))[1];
            float lo[8] = {f0.x, f0.y, f0.z, f0.w, f1.x, f1.y, f1.z, f1.w};
            float hi[8] = {f2.x, f2.y, f2.z, f2.w, f3.x, f3.y, f3.z, f3.w};
            v16bf a;
#pragma unroll
            for (int j = 0; j < 8; ++j) {
                float u = lo[j], w = hi[j];
                if (MODE == 1) {
                    u += bias[k0 + half * 8 + j];
                    u = u > 0.0f ? u : (__expf(u) - 1.0f);      // ELU fused
                    w += bias[k0 + 16 + half * 8 + j];
                    w = w > 0.0f ? w : (__expf(w) - 1.0f);
                }
                a[j]     = (__bf16)u;
                a[8 + j] = (__bf16)w;
            }
            // ---- 4 n-tiles reuse the A fragment ----
#pragma unroll
            for (int nt = 0; nt < 4; ++nt) {
                const v16bf b = *(const v16bf*)&Wl[(kk * 4 + nt) * 512 + lane * 16];
                acc[nt] = __builtin_amdgcn_wmma_f32_16x16x32_bf16(
                              false, a, false, b, (short)0, acc[nt], false, false);
            }
        }
    }

    // ---- store: VGPR i -> row m0 + half*8 + i, col nt*16 + lane%16 ----
#pragma unroll
    for (int i = 0; i < 8; ++i) {
        const long long r = m0 + half * 8 + i;
        if (r < Nrows) {
#pragma unroll
            for (int nt = 0; nt < 4; ++nt) {
                C[(size_t)r * 64 + nt * 16 + l16] = acc[nt][i];
            }
        }
    }
}

// ---------- per-(node,head) attention logits, layer 1 (H=8, D=8) ----------
__global__ void alpha1_k(const float* __restrict__ h1,
                         const float* __restrict__ a_src,
                         const float* __restrict__ a_dst,
                         float* __restrict__ as1, float* __restrict__ ad1,
                         long long NH) {
    long long t = (long long)blockIdx.x * blockDim.x + threadIdx.x;
    if (t >= NH) return;
    const long long n = t >> 3;
    const int h = (int)(t & 7);
    const float* hp = h1 + n * 64 + h * 8;
    float s = 0.f, d = 0.f;
#pragma unroll
    for (int j = 0; j < 8; ++j) {
        const float hv = hp[j];
        s += hv * a_src[h * 8 + j];
        d += hv * a_dst[h * 8 + j];
    }
    as1[t] = s;
    ad1[t] = d;
}

// ---------- per-node attention logits, layer 2 (H=1, D=64) ----------
__global__ void alpha2_k(const float* __restrict__ h2,
                         const float* __restrict__ a_src,
                         const float* __restrict__ a_dst,
                         float* __restrict__ as2, float* __restrict__ ad2,
                         long long N) {
    long long n = (long long)blockIdx.x * blockDim.x + threadIdx.x;
    if (n >= N) return;
    const float* hp = h2 + n * 64;
    float s = 0.f, d = 0.f;
#pragma unroll 8
    for (int j = 0; j < 64; ++j) {
        const float hv = hp[j];
        s += hv * a_src[j];
        d += hv * a_dst[j];
    }
    as2[n] = s;
    ad2[n] = d;
}

__device__ __forceinline__ void edge_ids(const long long* __restrict__ ei,
                                         long long idx, long long E,
                                         long long& s, long long& d) {
    if (idx < E) { s = ei[idx]; d = ei[E + idx]; }
    else         { s = d = idx - E; }           // appended self-loop
}

// ---------- edge pass 1: segment max of leaky-relu logits ----------
__global__ void edge_max_h8(const long long* __restrict__ ei, long long E, long long ET,
                            const float* __restrict__ as, const float* __restrict__ ad,
                            float* __restrict__ m) {
    long long idx = (long long)blockIdx.x * blockDim.x + threadIdx.x;
    if (idx >= ET) return;
    long long s, d; edge_ids(ei, idx, E, s, d);
#pragma unroll
    for (int h = 0; h < 8; ++h) {
        float e = as[s * 8 + h] + ad[d * 8 + h];
        e = e > 0.f ? e : 0.2f * e;
        atomicMaxF(&m[d * 8 + h], e);
    }
}

// ---------- edge pass 2: segment sum of exp(e - max) ----------
__global__ void edge_sum_h8(const long long* __restrict__ ei, long long E, long long ET,
                            const float* __restrict__ as, const float* __restrict__ ad,
                            const float* __restrict__ m, float* __restrict__ den) {
    long long idx = (long long)blockIdx.x * blockDim.x + threadIdx.x;
    if (idx >= ET) return;
    long long s, d; edge_ids(ei, idx, E, s, d);
#pragma unroll
    for (int h = 0; h < 8; ++h) {
        float e = as[s * 8 + h] + ad[d * 8 + h];
        e = e > 0.f ? e : 0.2f * e;
        atomicAdd(&den[d * 8 + h], __expf(e - m[d * 8 + h]));
    }
}

// ---------- edge pass 3: alpha * h[src] scattered into agg[dst] ----------
__global__ void edge_agg_h8(const long long* __restrict__ ei, long long E, long long ET,
                            const float* __restrict__ as, const float* __restrict__ ad,
                            const float* __restrict__ m, const float* __restrict__ den,
                            const float* __restrict__ h1, float* __restrict__ agg) {
    long long idx = (long long)blockIdx.x * blockDim.x + threadIdx.x;
    if (idx >= ET) return;
    long long s, d; edge_ids(ei, idx, E, s, d);
#pragma unroll
    for (int h = 0; h < 8; ++h) {
        float e = as[s * 8 + h] + ad[d * 8 + h];
        e = e > 0.f ? e : 0.2f * e;
        const float alpha = __expf(e - m[d * 8 + h]) / (den[d * 8 + h] + 1e-16f);
        const float4* hp = (const float4*)(h1 + s * 64 + h * 8);
        float* ap = agg + d * 64 + h * 8;
        const float4 p0 = hp[0], p1 = hp[1];
        atomicAdd(ap + 0, alpha * p0.x); atomicAdd(ap + 1, alpha * p0.y);
        atomicAdd(ap + 2, alpha * p0.z); atomicAdd(ap + 3, alpha * p0.w);
        atomicAdd(ap + 4, alpha * p1.x); atomicAdd(ap + 5, alpha * p1.y);
        atomicAdd(ap + 6, alpha * p1.z); atomicAdd(ap + 7, alpha * p1.w);
    }
}

// ---------- layer-2 edge passes (H=1, D=64) ----------
__global__ void edge_max_h1(const long long* __restrict__ ei, long long E, long long ET,
                            const float* __restrict__ as, const float* __restrict__ ad,
                            float* __restrict__ m) {
    long long idx = (long long)blockIdx.x * blockDim.x + threadIdx.x;
    if (idx >= ET) return;
    long long s, d; edge_ids(ei, idx, E, s, d);
    float e = as[s] + ad[d];
    e = e > 0.f ? e : 0.2f * e;
    atomicMaxF(&m[d], e);
}

__global__ void edge_sum_h1(const long long* __restrict__ ei, long long E, long long ET,
                            const float* __restrict__ as, const float* __restrict__ ad,
                            const float* __restrict__ m, float* __restrict__ den) {
    long long idx = (long long)blockIdx.x * blockDim.x + threadIdx.x;
    if (idx >= ET) return;
    long long s, d; edge_ids(ei, idx, E, s, d);
    float e = as[s] + ad[d];
    e = e > 0.f ? e : 0.2f * e;
    atomicAdd(&den[d], __expf(e - m[d]));
}

__global__ void edge_agg_h1(const long long* __restrict__ ei, long long E, long long ET,
                            const float* __restrict__ as, const float* __restrict__ ad,
                            const float* __restrict__ m, const float* __restrict__ den,
                            const float* __restrict__ h2, float* __restrict__ agg) {
    long long idx = (long long)blockIdx.x * blockDim.x + threadIdx.x;
    if (idx >= ET) return;
    long long s, d; edge_ids(ei, idx, E, s, d);
    float e = as[s] + ad[d];
    e = e > 0.f ? e : 0.2f * e;
    const float alpha = __expf(e - m[d]) / (den[d] + 1e-16f);
    const float4* hp = (const float4*)(h2 + s * 64);
    float* ap = agg + d * 64;
#pragma unroll
    for (int j = 0; j < 16; ++j) {
        const float4 p = hp[j];
        atomicAdd(ap + 4 * j + 0, alpha * p.x);
        atomicAdd(ap + 4 * j + 1, alpha * p.y);
        atomicAdd(ap + 4 * j + 2, alpha * p.z);
        atomicAdd(ap + 4 * j + 3, alpha * p.w);
    }
}

// ---------- final: out = log_softmax(agg2 + b2) ----------
__global__ void final_logsoftmax(const float* __restrict__ agg2,
                                 const float* __restrict__ b2,
                                 float* __restrict__ out, long long N) {
    long long n = (long long)blockIdx.x * blockDim.x + threadIdx.x;
    if (n >= N) return;
    float v[64];
    float mx = -__builtin_inff();
#pragma unroll 8
    for (int i = 0; i < 64; ++i) {
        v[i] = agg2[n * 64 + i] + b2[i];
        mx = fmaxf(mx, v[i]);
    }
    float ssum = 0.f;
#pragma unroll 8
    for (int i = 0; i < 64; ++i) ssum += __expf(v[i] - mx);
    const float l = mx + __logf(ssum);
#pragma unroll 8
    for (int i = 0; i < 64; ++i) out[n * 64 + i] = v[i] - l;
}

static inline unsigned gblk(long long n) { return (unsigned)((n + TB - 1) / TB); }

extern "C" void kernel_launch(void* const* d_in, const int* in_sizes, int n_in,
                              void* d_out, int out_size, void* d_ws, size_t ws_size,
                              hipStream_t stream) {
    const float*     x      = (const float*)d_in[0];
    const long long* ei     = (const long long*)d_in[1];   // int64 [2,E]
    const float*     W1     = (const float*)d_in[2];
    const float*     a_src1 = (const float*)d_in[3];
    const float*     a_dst1 = (const float*)d_in[4];
    const float*     b1     = (const float*)d_in[5];
    const float*     W2     = (const float*)d_in[6];
    const float*     a_src2 = (const float*)d_in[7];
    const float*     a_dst2 = (const float*)d_in[8];
    const float*     b2     = (const float*)d_in[9];

    const long long N  = in_sizes[0] / 512;   // F_IN = 512
    const long long E  = in_sizes[1] / 2;
    const long long ET = E + N;               // + self-loops

    // workspace layout (floats); N*292 floats total (~117 MB for N=100k)
    float* ws     = (float*)d_ws;
    float* h1     = ws;                 // N*64
    float* as1    = h1  + N * 64;       // N*8
    float* ad1    = as1 + N * 8;        // N*8
    float* h2     = ad1 + N * 8;        // N*64
    float* as2    = h2  + N * 64;       // N
    float* ad2    = as2 + N;            // N
    float* m1     = ad2 + N;            // N*8  --+ contiguous -inf region (N*9)
    float* m2     = m1  + N * 8;        // N    --+
    float* den1   = m2  + N;            // N*8  --+ contiguous zero region (N*137)
    float* den2   = den1 + N * 8;       // N      |
    float* agg1   = den2 + N;           // N*64   |
    float* agg2   = agg1 + N * 64;      // N*64 --+

    // re-init every call (atomics accumulate; replays must be deterministic)
    fill_f32<<<gblk(N * 9),   TB, 0, stream>>>(m1,   -__builtin_inff(), N * 9);
    fill_f32<<<gblk(N * 137), TB, 0, stream>>>(den1, 0.0f,              N * 137);

    const unsigned gemm_blocks = (unsigned)((N + 127) / 128);

    // ---- layer 1 ----
    gemm_wmma_n64<512, 0><<<gemm_blocks, TB, 0, stream>>>(x, W1, nullptr, h1, N);
    alpha1_k<<<gblk(N * 8), TB, 0, stream>>>(h1, a_src1, a_dst1, as1, ad1, N * 8);
    edge_max_h8<<<gblk(ET), TB, 0, stream>>>(ei, E, ET, as1, ad1, m1);
    edge_sum_h8<<<gblk(ET), TB, 0, stream>>>(ei, E, ET, as1, ad1, m1, den1);
    edge_agg_h8<<<gblk(ET), TB, 0, stream>>>(ei, E, ET, as1, ad1, m1, den1, h1, agg1);

    // ---- layer 2 (bias-add + ELU fused into GEMM A-operand load) ----
    gemm_wmma_n64<64, 1><<<gemm_blocks, TB, 0, stream>>>(agg1, W2, b1, h2, N);
    alpha2_k<<<gblk(N), TB, 0, stream>>>(h2, a_src2, a_dst2, as2, ad2, N);
    edge_max_h1<<<gblk(ET), TB, 0, stream>>>(ei, E, ET, as2, ad2, m2);
    edge_sum_h1<<<gblk(ET), TB, 0, stream>>>(ei, E, ET, as2, ad2, m2, den2);
    edge_agg_h1<<<gblk(ET), TB, 0, stream>>>(ei, E, ET, as2, ad2, m2, den2, h2, agg2);

    // ---- bias + log_softmax ----
    final_logsoftmax<<<gblk(N), TB, 0, stream>>>(agg2, b2, (float*)d_out, N);
}
